// Fre_EncoderLayer_19980187861721
// MI455X (gfx1250) — matrix operand order repaired and checked by reading, not compiled, not run on hardware
//
#include <hip/hip_runtime.h>

#define DEVFN __device__ __forceinline__

typedef __attribute__((ext_vector_type(16))) __bf16 v16bf;
typedef __attribute__((ext_vector_type(8)))  __bf16 v8bf;
typedef __attribute__((ext_vector_type(8)))  float  v8f;

DEVFN __bf16 f2bf(float f) {
  unsigned u = __builtin_bit_cast(unsigned, f);
  unsigned r = u + 0x7fffu + ((u >> 16) & 1u);
  unsigned short h = (unsigned short)(r >> 16);
  return __builtin_bit_cast(__bf16, h);
}
DEVFN float bf2f(__bf16 b) {
  unsigned short s = __builtin_bit_cast(unsigned short, b);
  unsigned u = ((unsigned)s) << 16;
  return __builtin_bit_cast(float, u);
}

// ---------------------------------------------------------------------------
// CDNA5 async global->LDS copy (ASYNCcnt-tracked DMA, no VGPR round-trip).
// VDST = per-lane LDS byte address (low 32 bits of flat pointer), VADDR = VA.
// ---------------------------------------------------------------------------
DEVFN void async_b128(void* lds_ptr, const void* gptr) {
  unsigned lds_off = (unsigned)(size_t)lds_ptr; // LDS byte address = low 32 bits of pointer
  asm volatile("global_load_async_to_lds_b128 %0, %1, off"
               :: "v"(lds_off), "v"(gptr)
               : "memory");
}
DEVFN void wait_async0() {
  asm volatile("s_wait_asynccnt 0x0" ::: "memory");
}

// A fragment: 16x32 bf16 (M x K). lane m = lane&15, K groups per ISA layout:
// VGPR0-3 hold K = kb..kb+7, VGPR4-7 hold K = 16+kb..16+kb+7, kb = (lane>>4)*8.
DEVFN v16bf load_frag_A(const __bf16* base, int ld, int lane) {
  const int m  = lane & 15;
  const int kb = (lane >> 4) << 3;
  const __bf16* p = base + (size_t)m * ld + kb;
  v8bf lo = *(const v8bf*)p;
  v8bf hi = *(const v8bf*)(p + 16);
  return __builtin_shufflevector(lo, hi, 0,1,2,3,4,5,6,7,8,9,10,11,12,13,14,15);
}
// B fragment: 32x16 bf16 (K x N). lane n = lane&15, K order same as A:
// elems 0-7 hold K = kb..kb+7, elems 8-15 hold K = 16+kb..16+kb+7, kb = (lane>>4)*8.
DEVFN v16bf load_frag_B(const __bf16* base, int ld, int lane) {
  const int n  = lane & 15;
  const int kb = (lane >> 4) << 3;
  v16bf r;
#pragma unroll
  for (int s = 0; s < 16; ++s) {
    const int k = (s < 8) ? (kb + s) : (16 + kb + s - 8);
    r[s] = base[(size_t)k * ld + n];
  }
  return r;
}

DEVFN v8f wmma_bf16(v16bf a, v16bf b, v8f c) {
  return __builtin_amdgcn_wmma_f32_16x16x32_bf16(false, a, false, b, (short)0, c,
                                                 false, false);
}

// ---------------------------------------------------------------------------
// Haar DWT along last axis: x (B,F,1024) f32 -> x1,x2 (B,F,512) bf16
// ---------------------------------------------------------------------------
__global__ __launch_bounds__(256) void dwt_kernel(const float* __restrict__ x,
                                                  __bf16* __restrict__ x1,
                                                  __bf16* __restrict__ x2) {
  const float INVSQRT2 = 0.70710678118654752440f;
  size_t idx = (size_t)blockIdx.x * 256 + threadIdx.x; // B*F*HALF = 4194304
  float a = x[2 * idx];
  float b = x[2 * idx + 1];
  x1[idx] = f2bf((a + b) * INVSQRT2);
  x2[idx] = f2bf((a - b) * INVSQRT2);
}

// ---------------------------------------------------------------------------
// Weight prep: w (Cout, Cin, 3) f32 -> wT (3, Cin, Cout) bf16 (K-major for B)
// ---------------------------------------------------------------------------
__global__ __launch_bounds__(256) void wprep_kernel(const float* __restrict__ w,
                                                    __bf16* __restrict__ wT,
                                                    int Cout, int Cin) {
  int idx = blockIdx.x * 256 + threadIdx.x;
  int total = Cout * Cin * 3;
  if (idx >= total) return;
  int co  = idx / (Cin * 3);
  int rem = idx - co * Cin * 3;
  int ci  = rem / 3;
  int t   = rem - ci * 3;
  wT[((size_t)t * Cin + ci) * Cout + co] = f2bf(w[idx]);
}

// ---------------------------------------------------------------------------
// Circular k=3 conv as WMMA GEMM.
// act: (B, L, Cin) bf16, wT: (3, Cin, Cout) bf16, bias: (Cout) f32
// Block tile: 128 (L) x 64 (Cout); 8 waves, each 16x64 (4 C-fragments).
// LDS staged via global_load_async_to_lds_b128 (ASYNCcnt protocol).
// OUTMODE: 0 = bf16 (B,L,Cout); 1 = bf16 transposed (B,Cout,L); 2 = f32 (B,L,Cout)
// ---------------------------------------------------------------------------
template <int OUTMODE, bool RELU>
__global__ __launch_bounds__(256) void conv3_wmma(
    const __bf16* __restrict__ act, const __bf16* __restrict__ wT,
    const float* __restrict__ bias, void* __restrict__ outp,
    int L, int Cin, int Cout) {
  const int tid = threadIdx.x, wave = tid >> 5, lane = tid & 31;
  const int m0 = blockIdx.x * 128, n0 = blockIdx.y * 64, b = blockIdx.z;
  const int Lm = L - 1; // L is a power of two

  __shared__ __bf16 ldsA[130 * 40];     // 130 circular rows x 32 ch (pad 40)
  __shared__ __bf16 ldsB[3 * 32 * 72];  // 3 taps x 32 k x 64 n (pad 72)

  v8f acc[4] = {};
  const __bf16* actB = act + (size_t)b * L * Cin;

  for (int kc = 0; kc < Cin; kc += 32) {
    __syncthreads();
    // Stage A: rows m0-1 .. m0+128 (circular), channels [kc, kc+32)
    for (int i = tid; i < 130 * 4; i += 256) {
      int r = i >> 2, cc = (i & 3) << 3;
      int src = (m0 - 1 + r + L) & Lm;
      async_b128(ldsA + r * 40 + cc, actB + (size_t)src * Cin + kc + cc);
    }
    // Stage B: 3 taps of wT[t][kc..kc+31][n0..n0+63]
    for (int i = tid; i < 3 * 32 * 8; i += 256) {
      int t = i >> 8;
      int rr = i & 255, kk = rr >> 3, cc = (rr & 7) << 3;
      async_b128(ldsB + (t * 32 + kk) * 72 + cc,
                 wT + ((size_t)t * Cin + kc + kk) * Cout + n0 + cc);
    }
    // Prefetch next K-chunk of activations into cache while this chunk lands.
    if (kc + 32 < Cin && tid < 130) {
      int src = (m0 - 1 + tid + L) & Lm;
      __builtin_prefetch(actB + (size_t)src * Cin + kc + 32, 0, 3);
    }
    wait_async0();
    __syncthreads();

#pragma unroll
    for (int t = 0; t < 3; ++t) {
      v16bf a = load_frag_A(ldsA + (wave * 16 + t) * 40, 40, lane);
#pragma unroll
      for (int nt = 0; nt < 4; ++nt) {
        v16bf bf = load_frag_B(ldsB + t * 32 * 72 + nt * 16, 72, lane);
        acc[nt] = wmma_bf16(a, bf, acc[nt]);
      }
    }
  }

  const int n = lane & 15, mh = (lane >> 4) << 3;
  const int mrow = m0 + wave * 16 + mh;
#pragma unroll
  for (int nt = 0; nt < 4; ++nt) {
    const int nglob = n0 + nt * 16 + n;
    const float bb = bias[nglob];
    float r[8];
#pragma unroll
    for (int j = 0; j < 8; ++j) {
      float y = acc[nt][j] + bb;
      if (RELU) y = fmaxf(y, 0.0f);
      r[j] = y;
    }
    if (OUTMODE == 0) {
      __bf16* out = (__bf16*)outp;
#pragma unroll
      for (int j = 0; j < 8; ++j)
        out[((size_t)b * L + mrow + j) * Cout + nglob] = f2bf(r[j]);
    } else if (OUTMODE == 1) {
      __bf16* out = (__bf16*)outp;
      v8bf o;
#pragma unroll
      for (int j = 0; j < 8; ++j) o[j] = f2bf(r[j]);
      *(v8bf*)(out + ((size_t)b * Cout + nglob) * L + mrow) = o;
    } else {
      float* out = (float*)outp;
#pragma unroll
      for (int j = 0; j < 8; ++j)
        out[((size_t)b * L + mrow + j) * Cout + nglob] = r[j];
    }
  }
}

// ---------------------------------------------------------------------------
// Attention scores: S = scale * q . k^T per (b,h). q,k: (B,512,1024) bf16,
// head h at channels [h*64,(h+1)*64). probs: (B*H, 512, 512) bf16.
// Grid (F/128, F/64, B*H).
// ---------------------------------------------------------------------------
__global__ __launch_bounds__(256) void scores_wmma(
    const __bf16* __restrict__ q, const __bf16* __restrict__ k,
    __bf16* __restrict__ probs, float scale) {
  const int tid = threadIdx.x, wave = tid >> 5, lane = tid & 31;
  const int m0 = blockIdx.x * 128, n0 = blockIdx.y * 64;
  const int bh = blockIdx.z, b = bh >> 4, h = bh & 15;

  __shared__ __bf16 ldsK[64 * 72]; // K^T tile: [d][key] with pad
  for (int i = tid; i < 64 * 8; i += 256) {
    int n = i >> 3, dc = (i & 7) << 3;
    v8bf v = *(const v8bf*)(k + ((size_t)b * 512 + n0 + n) * 1024 + h * 64 + dc);
#pragma unroll
    for (int j = 0; j < 8; ++j) ldsK[(dc + j) * 72 + n] = v[j];
  }
  __syncthreads();

  v8f acc[4] = {};
  const __bf16* qb = q + ((size_t)b * 512 + m0 + wave * 16) * 1024 + h * 64;
#pragma unroll
  for (int kc = 0; kc < 64; kc += 32) {
    v16bf a = load_frag_A(qb + kc, 1024, lane);
#pragma unroll
    for (int nt = 0; nt < 4; ++nt) {
      v16bf bf = load_frag_B(ldsK + kc * 72 + nt * 16, 72, lane);
      acc[nt] = wmma_bf16(a, bf, acc[nt]);
    }
  }
  const int n = lane & 15, mh = (lane >> 4) << 3;
  const int mrow = m0 + wave * 16 + mh;
#pragma unroll
  for (int nt = 0; nt < 4; ++nt)
#pragma unroll
    for (int j = 0; j < 8; ++j)
      probs[((size_t)bh * 512 + mrow + j) * 512 + n0 + nt * 16 + n] =
          f2bf(acc[nt][j] * scale);
}

// ---------------------------------------------------------------------------
// Row softmax over 512 (bf16 in/out). One block per row, 2 elems/thread.
// ---------------------------------------------------------------------------
__global__ __launch_bounds__(256) void softmax_rows(__bf16* __restrict__ p) {
  __shared__ float redm[8];
  __shared__ float reds[8];
  const int tid = threadIdx.x;
  __bf16* r = p + (size_t)blockIdx.x * 512;
  float v0 = bf2f(r[tid]), v1 = bf2f(r[tid + 256]);
  float m = fmaxf(v0, v1);
#pragma unroll
  for (int o = 16; o > 0; o >>= 1) m = fmaxf(m, __shfl_xor(m, o));
  if ((tid & 31) == 0) redm[tid >> 5] = m;
  __syncthreads();
  float M = redm[0];
#pragma unroll
  for (int i = 1; i < 8; ++i) M = fmaxf(M, redm[i]);
  float e0 = __expf(v0 - M), e1 = __expf(v1 - M);
  float s = e0 + e1;
#pragma unroll
  for (int o = 16; o > 0; o >>= 1) s += __shfl_xor(s, o);
  if ((tid & 31) == 0) reds[tid >> 5] = s;
  __syncthreads();
  float S = 0.f;
#pragma unroll
  for (int i = 0; i < 8; ++i) S += reds[i];
  float inv = 1.0f / S;
  r[tid]       = f2bf(e0 * inv);
  r[tid + 256] = f2bf(e1 * inv);
}

// ---------------------------------------------------------------------------
// ctx = P @ V per (b,h), stored transposed directly into the uns-conv input:
// unsin (B, 1024(dmodel), 1024(2F)) at [b][h*64+d][branch*512+f].
// Grid (F/128, 1, B*H).
// ---------------------------------------------------------------------------
__global__ __launch_bounds__(256) void pv_wmma(
    const __bf16* __restrict__ probs, const __bf16* __restrict__ v,
    __bf16* __restrict__ unsin, int branch) {
  const int tid = threadIdx.x, wave = tid >> 5, lane = tid & 31;
  const int m0 = blockIdx.x * 128;
  const int bh = blockIdx.z, b = bh >> 4, h = bh & 15;

  v8f acc[4] = {};
  const __bf16* pb = probs + ((size_t)bh * 512 + m0 + wave * 16) * 512;
  const __bf16* vb = v + (size_t)b * 512 * 1024 + h * 64;
  for (int kc = 0; kc < 512; kc += 32) {
    v16bf a = load_frag_A(pb + kc, 512, lane);
#pragma unroll
    for (int nt = 0; nt < 4; ++nt) {
      v16bf bf = load_frag_B(vb + (size_t)kc * 1024 + nt * 16, 1024, lane);
      acc[nt] = wmma_bf16(a, bf, acc[nt]);
    }
  }
  const int n = lane & 15, mh = (lane >> 4) << 3;
  const int mrow = m0 + wave * 16 + mh;
#pragma unroll
  for (int nt = 0; nt < 4; ++nt) {
    v8bf o;
#pragma unroll
    for (int j = 0; j < 8; ++j) o[j] = f2bf(acc[nt][j]);
    *(v8bf*)(unsin + ((size_t)b * 1024 + h * 64 + nt * 16 + n) * 1024 +
             branch * 512 + mrow) = o;
  }
}

// ---------------------------------------------------------------------------
// LayerNorm over last axis C (per row). FINAL: add residual, write f32 d_out.
// Otherwise write bf16 for the next conv.
// ---------------------------------------------------------------------------
template <bool FINAL>
__global__ __launch_bounds__(256) void ln_rows(
    const float* __restrict__ in, const float* __restrict__ g,
    const float* __restrict__ be, const float* __restrict__ resid,
    void* __restrict__ outp, int C) {
  __shared__ float rs[8];
  __shared__ float rs2[8];
  const int tid = threadIdx.x;
  const size_t row = blockIdx.x;
  const float* r = in + row * C;
  float s = 0.f, s2 = 0.f;
  for (int i = tid; i < C; i += 256) {
    float xv = r[i];
    s += xv;
    s2 += xv * xv;
  }
#pragma unroll
  for (int o = 16; o > 0; o >>= 1) {
    s += __shfl_xor(s, o);
    s2 += __shfl_xor(s2, o);
  }
  if ((tid & 31) == 0) { rs[tid >> 5] = s; rs2[tid >> 5] = s2; }
  __syncthreads();
  float S = 0.f, S2 = 0.f;
#pragma unroll
  for (int i = 0; i < 8; ++i) { S += rs[i]; S2 += rs2[i]; }
  const float mean = S / C;
  const float var = S2 / C - mean * mean;
  const float rstd = rsqrtf(var + 1e-5f);
  for (int i = tid; i < C; i += 256) {
    float y = (r[i] - mean) * rstd * g[i] + be[i];
    if (FINAL)
      ((float*)outp)[row * C + i] = y + resid[row * C + i];
    else
      ((__bf16*)outp)[row * C + i] = f2bf(y);
  }
}

// ---------------------------------------------------------------------------
extern "C" void kernel_launch(void* const* d_in, const int* in_sizes, int n_in,
                              void* d_out, int out_size, void* d_ws,
                              size_t ws_size, hipStream_t stream) {
  (void)in_sizes; (void)n_in; (void)out_size; (void)ws_size;
  const float* x      = (const float*)d_in[0];
  const float* Wq     = (const float*)d_in[1];
  const float* bq     = (const float*)d_in[2];
  const float* Wk     = (const float*)d_in[3];
  const float* bk     = (const float*)d_in[4];
  const float* Wv     = (const float*)d_in[5];
  const float* bv     = (const float*)d_in[6];
  const float* uns_w  = (const float*)d_in[7];
  const float* uns_b  = (const float*)d_in[8];
  const float* cv_w1  = (const float*)d_in[9];
  const float* cv_b1  = (const float*)d_in[10];
  const float* cv_w2  = (const float*)d_in[11];
  const float* cv_b2  = (const float*)d_in[12];
  const float* cv_g   = (const float*)d_in[13];
  const float* cv_be  = (const float*)d_in[14];
  const float* up_w1  = (const float*)d_in[15];
  const float* up_b1  = (const float*)d_in[16];
  const float* up_w2  = (const float*)d_in[17];
  const float* up_b2  = (const float*)d_in[18];
  const float* up_g   = (const float*)d_in[19];
  const float* up_be  = (const float*)d_in[20];

  const size_t B = 16, F = 512, DM = 1024, HC = 512, DFF = 2048, H = 16;

  char* p = (char*)d_ws;
  auto alloc = [&](size_t bytes) -> char* {
    char* r = p;
    p += (bytes + 255) & ~(size_t)255;
    return r;
  };
  __bf16* x1    = (__bf16*)alloc(B * F * HC * 2);
  __bf16* x2    = (__bf16*)alloc(B * F * HC * 2);
  __bf16* wqT   = (__bf16*)alloc(3 * HC * DM * 2);
  __bf16* wkT   = (__bf16*)alloc(3 * HC * DM * 2);
  __bf16* wvT   = (__bf16*)alloc(3 * HC * DM * 2);
  __bf16* unsT  = (__bf16*)alloc(3 * DM * HC * 2);
  __bf16* cv1T  = (__bf16*)alloc(3 * DM * DFF * 2);
  __bf16* cv2T  = (__bf16*)alloc(3 * DFF * HC * 2);
  __bf16* up1T  = (__bf16*)alloc(3 * HC * DFF * 2);
  __bf16* up2T  = (__bf16*)alloc(3 * DFF * DM * 2);
  __bf16* q1    = (__bf16*)alloc(B * F * DM * 2);
  __bf16* k1    = (__bf16*)alloc(B * F * DM * 2);
  __bf16* v1    = (__bf16*)alloc(B * F * DM * 2);
  __bf16* q2    = (__bf16*)alloc(B * F * DM * 2);
  __bf16* k2    = (__bf16*)alloc(B * F * DM * 2);
  __bf16* v2    = (__bf16*)alloc(B * F * DM * 2);
  __bf16* probs = (__bf16*)alloc(B * H * F * F * 2);
  __bf16* unsin = (__bf16*)alloc(B * DM * (2 * F) * 2);
  __bf16* cvin  = (__bf16*)alloc(B * F * DM * 2);
  __bf16* cv1o  = (__bf16*)alloc(B * F * DFF * 2);
  float*  cv2o  = (float*)alloc(B * F * HC * 4);
  __bf16* upin  = (__bf16*)alloc(B * F * HC * 2);
  __bf16* up1o  = (__bf16*)alloc(B * F * DFF * 2);
  float*  up2o  = (float*)alloc(B * F * DM * 4);

  const dim3 blk(256);

  // 1) DWT
  dwt_kernel<<<dim3((unsigned)(B * F * HC / 256)), blk, 0, stream>>>(x, x1, x2);

  // 2) Weight prep (fp32 -> bf16, (Cout,Cin,3) -> (3,Cin,Cout))
  auto prep = [&](const float* w, __bf16* wt, int Cout, int Cin) {
    int total = Cout * Cin * 3;
    wprep_kernel<<<dim3((total + 255) / 256), blk, 0, stream>>>(w, wt, Cout, Cin);
  };
  prep(Wq, wqT, 1024, 512);
  prep(Wk, wkT, 1024, 512);
  prep(Wv, wvT, 1024, 512);
  prep(uns_w, unsT, 512, 1024);
  prep(cv_w1, cv1T, 2048, 1024);
  prep(cv_w2, cv2T, 512, 2048);
  prep(up_w1, up1T, 2048, 512);
  prep(up_w2, up2T, 1024, 2048);

  // 3) Projections (x1 -> qh,kh,vh ; x2 -> ql,kl,vl)
  conv3_wmma<0, false><<<dim3(4, 16, 16), blk, 0, stream>>>(x1, wqT, bq, q1, 512, 512, 1024);
  conv3_wmma<0, false><<<dim3(4, 16, 16), blk, 0, stream>>>(x1, wkT, bk, k1, 512, 512, 1024);
  conv3_wmma<0, false><<<dim3(4, 16, 16), blk, 0, stream>>>(x1, wvT, bv, v1, 512, 512, 1024);
  conv3_wmma<0, false><<<dim3(4, 16, 16), blk, 0, stream>>>(x2, wqT, bq, q2, 512, 512, 1024);
  conv3_wmma<0, false><<<dim3(4, 16, 16), blk, 0, stream>>>(x2, wkT, bk, k2, 512, 512, 1024);
  conv3_wmma<0, false><<<dim3(4, 16, 16), blk, 0, stream>>>(x2, wvT, bv, v2, 512, 512, 1024);

  const float scale = 0.25f; // 1/sqrt(N_HEADS)

  // 4) Attention branch H: softmax(qh . kl^T) @ vh -> unsin[..][0:512]
  scores_wmma<<<dim3(4, 8, 256), blk, 0, stream>>>(q1, k2, probs, scale);
  softmax_rows<<<dim3((unsigned)(B * H * F)), blk, 0, stream>>>(probs);
  pv_wmma<<<dim3(4, 1, 256), blk, 0, stream>>>(probs, v1, unsin, 0);
  // Attention branch L: softmax(ql . kh^T) @ vl -> unsin[..][512:1024]
  scores_wmma<<<dim3(4, 8, 256), blk, 0, stream>>>(q2, k1, probs, scale);
  softmax_rows<<<dim3((unsigned)(B * H * F)), blk, 0, stream>>>(probs);
  pv_wmma<<<dim3(4, 1, 256), blk, 0, stream>>>(probs, v2, unsin, 1);

  // 5) uns conv (Cin=2F=1024 -> Cout=512 over L=1024), store transposed for cv1
  conv3_wmma<1, false><<<dim3(8, 8, 16), blk, 0, stream>>>(unsin, unsT, uns_b, cvin, 1024, 1024, 512);

  // 6) Convlayer: 1024 -> 2048 (ReLU) -> 512, LN
  conv3_wmma<0, true ><<<dim3(4, 32, 16), blk, 0, stream>>>(cvin, cv1T, cv_b1, cv1o, 512, 1024, 2048);
  conv3_wmma<2, false><<<dim3(4, 8, 16), blk, 0, stream>>>(cv1o, cv2T, cv_b2, cv2o, 512, 2048, 512);
  ln_rows<false><<<dim3((unsigned)(B * F)), blk, 0, stream>>>(cv2o, cv_g, cv_be, nullptr, upin, 512);

  // 7) upsizeConvlayer: 512 -> 2048 (ReLU) -> 1024, LN + residual
  conv3_wmma<0, true ><<<dim3(4, 32, 16), blk, 0, stream>>>(upin, up1T, up_b1, up1o, 512, 512, 2048);
  conv3_wmma<2, false><<<dim3(4, 16, 16), blk, 0, stream>>>(up1o, up2T, up_b2, up2o, 512, 2048, 1024);
  ln_rows<true ><<<dim3((unsigned)(B * F)), blk, 0, stream>>>(up2o, up_g, up_be, x, d_out, 1024);
}
